// MultiHeadAttention_72756745994310
// MI455X (gfx1250) — matrix operand order repaired and checked by reading, not compiled, not run on hardware
//
#include <hip/hip_runtime.h>
#include <hip/hip_fp16.h>

// Problem constants (fixed by the reference setup)
#define BB   8
#define QQ   1024
#define SS   2048
#define DD   1024
#define HH   16
#define DH   64          // head dim
#define QT   32          // Q rows per workgroup
#define NTHREADS 128     // 4 waves (wave32)
#define TINYF 1e-13f

#define EXS  2056        // padded row stride (floats) for score buffer: 8 mod 64 banks
#define VTS  40          // padded row stride (halfs) for V^T tile

typedef __attribute__((ext_vector_type(16))) _Float16 v16h;
typedef __attribute__((ext_vector_type(8)))  _Float16 v8h;
typedef __attribute__((ext_vector_type(8)))  float    v8f;
typedef __attribute__((ext_vector_type(4)))  float    f4;

// ---- dynamic LDS layout (bytes) ----
#define OFF_EX    0                       // 32*2056*4 = 263168
#define OFF_QT    263168                  // 32*64*2   =   4096
#define OFF_MK    267264                  // 2048*4    =   8192
#define OFF_VT    275456                  // 2*64*40*2 =  10240
#define OFF_RMAX  285696                  // 32*4
#define OFF_RINV  285824                  // 32*4
#define OFF_RED   285952                  // 32*4*4
#define SMEM_TOTAL 286464

__device__ __forceinline__ v8f wmma16(v16h a, v16h b, v8f c) {
  // D = A(16x32 f16) x B(32x16 f16) + C(16x16 f32)
  return __builtin_amdgcn_wmma_f32_16x16x32_f16(
      /*neg_a=*/false, a, /*neg_b=*/false, b,
      /*c_mod=*/(short)0, c, /*reuse_a=*/false, /*reuse_b=*/false);
}

// load 8 contiguous f32 at p (16B aligned), convert -> frag[base..base+7]
__device__ __forceinline__ void cvt_f32x8(v16h& f, int base, const float* p) {
  f4 a = *(const f4*)p;
  f4 b = *(const f4*)(p + 4);
  f[base + 0] = (_Float16)a.x; f[base + 1] = (_Float16)a.y;
  f[base + 2] = (_Float16)a.z; f[base + 3] = (_Float16)a.w;
  f[base + 4] = (_Float16)b.x; f[base + 5] = (_Float16)b.y;
  f[base + 6] = (_Float16)b.z; f[base + 7] = (_Float16)b.w;
}

// load 8 contiguous f16 at p (16B aligned) -> frag[base..base+7]
__device__ __forceinline__ void ld_h8(v16h& f, int base, const _Float16* p) {
  v8h x = *(const v8h*)p;
#pragma unroll
  for (int i = 0; i < 8; ++i) f[base + i] = x[i];
}

__global__ __launch_bounds__(NTHREADS)
void mha_fused_kernel(const float* __restrict__ Qp, const float* __restrict__ Kp,
                      const float* __restrict__ Vp, const int*   __restrict__ Mp,
                      float* __restrict__ Op, float* __restrict__ Ap) {
  extern __shared__ char smem[];
  float*    ex   = (float*)   (smem + OFF_EX);
  _Float16* qlds = (_Float16*)(smem + OFF_QT);
  float*    mlds = (float*)   (smem + OFF_MK);
  _Float16* vt   = (_Float16*)(smem + OFF_VT);   // two buffers of 64*VTS
  float*    rmax = (float*)   (smem + OFF_RMAX);
  float*    rinv = (float*)   (smem + OFF_RINV);
  float*    red  = (float*)   (smem + OFF_RED);

  const int tid  = threadIdx.x;
  const int lane = tid & 31;
  const int wv   = tid >> 5;            // wave id 0..3
  const int n16  = lane & 15;
  const int hi16 = (lane >= 16);
  const int off8 = hi16 ? 8 : 0;        // A-frag K offset
  const int offB = hi16 ? 16 : 0;       // B-frag K offset

  const int bid = blockIdx.x;
  const int qt  = bid & 31;             // Q tile within head
  const int h   = (bid >> 5) & 15;
  const int b   = bid >> 9;
  const int q0  = qt * QT;

  // ---------------- Phase 0: stage Q tile (scaled) + mask row ----------------
  {
    const int r = tid >> 2, g = tid & 3;               // 32 rows x 4 groups of 16
    const float* qp = Qp + ((size_t)b * QQ + q0 + r) * DD + h * DH + g * 16;
#pragma unroll
    for (int i = 0; i < 16; i += 4) {
      f4 x = *(const f4*)(qp + i);
      int c = g * 16 + i;
      qlds[r * DH + c + 0] = (_Float16)(x.x * 0.125f);
      qlds[r * DH + c + 1] = (_Float16)(x.y * 0.125f);
      qlds[r * DH + c + 2] = (_Float16)(x.z * 0.125f);
      qlds[r * DH + c + 3] = (_Float16)(x.w * 0.125f);
    }
    const int* mp = Mp + (size_t)b * SS + tid * 16;
#pragma unroll
    for (int i = 0; i < 16; ++i) mlds[tid * 16 + i] = (float)mp[i];
  }
  __syncthreads();

  // ---------------- Phase 1: scores = (Q/8) K^T, masked, -> LDS f32 ----------
  v16h aQ[2][2];
#pragma unroll
  for (int ms = 0; ms < 2; ++ms)
#pragma unroll
    for (int kc = 0; kc < 2; ++kc) {
      const _Float16* row = qlds + (ms * 16 + n16) * DH + kc * 32 + off8;
      ld_h8(aQ[ms][kc], 0, row);
      ld_h8(aQ[ms][kc], 8, row + 16);
    }

  for (int st = wv; st < SS / 16; st += 4) {
    const int scol = st * 16 + n16;
    const float* kp = Kp + ((size_t)b * SS + scol) * DD + h * DH;
    if (st + 4 < SS / 16)   // per-lane prefetch of next iteration's K data (L2 -> WGP$)
      __builtin_prefetch(kp + (size_t)64 * DD, 0, 0);
    v16h bK[2];
#pragma unroll
    for (int kc = 0; kc < 2; ++kc) {
      const float* p = kp + kc * 32 + off8;   // d values off8..+7 and off8+16..+23
      cvt_f32x8(bK[kc], 0, p);
      cvt_f32x8(bK[kc], 8, p + 16);
    }
    const float mval = mlds[scol];
#pragma unroll
    for (int ms = 0; ms < 2; ++ms) {
      v8f c = {};
      c = wmma16(aQ[ms][0], bK[0], c);
      c = wmma16(aQ[ms][1], bK[1], c);
      const int r0 = ms * 16 + (hi16 ? 8 : 0);
#pragma unroll
      for (int v = 0; v < 8; ++v)
        ex[(r0 + v) * EXS + scol] = c[v] * mval;   // scores * mask
    }
  }
  __syncthreads();

  // ---------------- Phase 2: masked softmax over S in LDS --------------------
  // lane-in-wave differs in column -> banks spread; rows rotate via EXS padding
  const int prow = wv * 8 + (lane >> 2);    // row 0..31
  const int psub = lane & 3;                // column interleave 0..3
  float* exrow = ex + prow * EXS;
  {
    float mx = -3.4e38f;
#pragma unroll 4
    for (int i = 0; i < 128; ++i) {
      f4 x = *(const f4*)(exrow + psub * 4 + i * 16);
      mx = fmaxf(mx, fmaxf(fmaxf(x.x, x.y), fmaxf(x.z, x.w)));
    }
    red[prow * 4 + psub] = mx;
  }
  __syncthreads();
  if (tid < 32)
    rmax[tid] = fmaxf(fmaxf(red[tid * 4], red[tid * 4 + 1]),
                      fmaxf(red[tid * 4 + 2], red[tid * 4 + 3]));
  __syncthreads();
  {
    const float m_ = rmax[prow];
    float sum = 0.f;
#pragma unroll 4
    for (int i = 0; i < 128; ++i) {
      const int c = psub * 4 + i * 16;
      f4 x = *(f4*)(exrow + c);
      f4 mm = *(const f4*)(mlds + c);
      x.x = __expf(x.x - m_) * mm.x; x.y = __expf(x.y - m_) * mm.y;
      x.z = __expf(x.z - m_) * mm.z; x.w = __expf(x.w - m_) * mm.w;
      sum += (x.x + x.y) + (x.z + x.w);
      *(f4*)(exrow + c) = x;
    }
    red[prow * 4 + psub] = sum;
  }
  __syncthreads();
  if (tid < 32) {
    float s4 = red[tid * 4] + red[tid * 4 + 1] + red[tid * 4 + 2] + red[tid * 4 + 3];
    rinv[tid] = 1.0f / (s4 + TINYF);
  }
  __syncthreads();
  // stream normalized attention to global: wave-contiguous 512B bursts, nontemporal
#pragma unroll 2
  for (int r8 = 0; r8 < 8; ++r8) {
    const int r = wv * 8 + r8;
    const float inv = rinv[r];
    const float* er = ex + r * EXS;
    f4* ag = (f4*)(Ap + (((size_t)b * QQ + q0 + r) * HH + h) * SS);
#pragma unroll 4
    for (int j = 0; j < 16; ++j) {
      const int c = lane * 4 + j * 128;
      f4 x = *(const f4*)(er + c);
      x.x *= inv; x.y *= inv; x.z *= inv; x.w *= inv;
      __builtin_nontemporal_store(x, ag + (c >> 2));
    }
  }
  __syncthreads();

  // ---------------- Phase 3: out_unnorm = exp_attn @ V (double-buffered V^T) -
  v8f acc[2] = {{}, {}};
  const int dv0 = wv * 16 + n16;            // each wave owns 16 of 64 dv cols
  const int sr = tid >> 2, g = tid & 3;     // V staging: 32 s-rows x 4 col groups
  f4 nxt[4];
  auto loadV = [&](int sc) {
    const float* vp = Vp + ((size_t)b * SS + sc * 32 + sr) * DD + h * DH + g * 16;
    nxt[0] = *(const f4*)(vp + 0);
    nxt[1] = *(const f4*)(vp + 4);
    nxt[2] = *(const f4*)(vp + 8);
    nxt[3] = *(const f4*)(vp + 12);
  };
  auto storeV = [&](_Float16* dst) {
#pragma unroll
    for (int i = 0; i < 16; ++i)
      dst[(g * 16 + i) * VTS + sr] = (_Float16)nxt[i >> 2][i & 3];
  };

  loadV(0);
  storeV(vt);
  __syncthreads();

  for (int sc = 0; sc < SS / 32; ++sc) {
    _Float16* cur = vt + (sc & 1) * (64 * VTS);
    if (sc + 1 < SS / 32) loadV(sc + 1);    // overlap next tile's loads with WMMA

    v16h bV;
    {
      const _Float16* pr = cur + dv0 * VTS + off8;
      ld_h8(bV, 0, pr);
      ld_h8(bV, 8, pr + 16);
    }
#pragma unroll
    for (int ms = 0; ms < 2; ++ms) {
      v16h aA;
      const float* er = ex + (ms * 16 + n16) * EXS + sc * 32 + off8;
      cvt_f32x8(aA, 0, er);
      cvt_f32x8(aA, 8, er + 16);
      acc[ms] = wmma16(aA, bV, acc[ms]);
    }
    if (sc + 1 < SS / 32) storeV(vt + ((sc + 1) & 1) * (64 * VTS));
    __syncthreads();
  }

  // ---------------- Phase 4: normalize + store out ---------------------------
#pragma unroll
  for (int ms = 0; ms < 2; ++ms) {
    const int r0 = ms * 16 + (hi16 ? 8 : 0);
#pragma unroll
    for (int v = 0; v < 8; ++v) {
      float o = acc[ms][v] * rinv[r0 + v];
      float* op = Op + ((size_t)b * QQ + q0 + r0 + v) * DD + h * DH + dv0;
      __builtin_nontemporal_store(o, op);
    }
  }
}

extern "C" void kernel_launch(void* const* d_in, const int* in_sizes, int n_in,
                              void* d_out, int out_size, void* d_ws, size_t ws_size,
                              hipStream_t stream) {
  const float* q  = (const float*)d_in[0];
  const float* k  = (const float*)d_in[1];
  const float* v  = (const float*)d_in[2];
  const int*   mk = (const int*)d_in[3];
  float* outp  = (float*)d_out;                       // [B,Q,D]
  float* attnp = outp + (size_t)BB * QQ * DD;         // [B,Q,H,S]

  dim3 grid(BB * HH * (QQ / QT));                     // 4096 workgroups
  mha_fused_kernel<<<grid, NTHREADS, SMEM_TOTAL, stream>>>(q, k, v, mk, outp, attnp);
}
